// GraphClassifier_85444079386804
// MI455X (gfx1250) — matrix operand order repaired and checked by reading, not compiled, not run on hardware
//
#include <hip/hip_runtime.h>
#include <hip/hip_bf16.h>

typedef __attribute__((ext_vector_type(16))) _Float16 v16h;
typedef __attribute__((ext_vector_type(8)))  _Float16 v8h;
typedef __attribute__((ext_vector_type(8)))  float    v8f;

// ---------------- problem constants ----------------
constexpr int NNODE = 65536;   // B*N
constexpr int KNEI  = 8;
constexpr int INPD  = 32;
constexpr int FEATD = 16;
constexpr int EMBD  = 32;
constexpr int OUTD  = 96;      // L*EMB
constexpr int BATCH = 512;
constexpr int NSEQ  = 128;
constexpr int NEDGE = 1048576; // B*N*DEG
constexpr int EPG   = 2048;    // N*DEG edges per graph
constexpr int NRREL = 201;     // NR+1
constexpr int PPATH = 64;      // K*K
constexpr int GREPD = 352;     // 3*OUT + 2*EMB

// ---------------- workspace layout (float units) ----------------
constexpr size_t OFF_HA    = 0;            // NN*32 f32   (later aliased by msg16)
constexpr size_t OFF_HB    = 2097152;      // NN*32 f32
constexpr size_t OFF_AGG   = 4194304;      // NN*32 f32
constexpr size_t OFF_DEG   = 6291456;      // NN f32
constexpr size_t OFF_RA    = 6356992;      // 201*32 f32
constexpr size_t OFF_RB    = 6363424;      // 201*32 f32
constexpr size_t OFF_REPR  = 6369856;      // NN*96 f32
constexpr size_t OFF_GI    = 12661312;     // NN*288 f32 (later aliased: node_hidden, path-last)
constexpr size_t OFF_YSF16 = 31535680;     // NN*96 f16  (3145728 floats)
constexpr size_t OFF_YSB16 = 34681408;     // NN*96 f16
constexpr size_t OFF_H0    = 37827136;     // B*96 f32
constexpr size_t OFF_HST   = 37876288;     // B*96 f32
constexpr size_t OFF_HST16 = 37925440;     // B*96 f16 (24576 floats)
constexpr size_t OFF_GH    = 37950016;     // B*288 f32
constexpr size_t OFF_GREP  = 38097472;     // B*352 f32
constexpr size_t OFF_WIH16 = 38277696;     // 288*96 f16 (13824 floats)
constexpr size_t OFF_WHH16 = 38291520;     // 288*96 f16
constexpr size_t OFF_WO16  = 38305344;     // 96*192 f16 (transposed, 9216 floats)
// msg16 aliases OFF_HA (NN*96 f16 = 3145728 floats <= h_a+h_b+agg region)

// ---------------- WMMA fragment helpers (wave32, CDNA5 layouts) ----------------
// A 16x32: lanes 0-15 row M=lane hold K {0..7,16..23}; lanes 16-31 hold K {8..15,24..31}
__device__ __forceinline__ v16h load_a_f32(const float* __restrict__ src, int lda,
                                           int m0, int k0, const float* __restrict__ rowscale,
                                           int lane) {
  int m  = m0 + (lane & 15);
  int bk = (lane >> 4) * 8;
  float s = rowscale ? rowscale[m] : 1.0f;
  const float* row = src + (size_t)m * lda + k0;
  v16h a;
#pragma unroll
  for (int e = 0; e < 16; ++e) {
    int k = ((e & 8) ? 16 : 0) + bk + (e & 7);
    a[e] = (_Float16)(row[k] * s);
  }
  return a;
}

// f16 A fragment: two aligned 16B vector loads per lane
__device__ __forceinline__ v16h load_a_f16(const _Float16* __restrict__ src, int lda,
                                           int m0, int k0, int lane) {
  int m  = m0 + (lane & 15);
  int bk = (lane >> 4) * 8;
  const _Float16* row = src + (size_t)m * lda + k0;
  v8h lo = *(const v8h*)(row + bk);
  v8h hi = *(const v8h*)(row + 16 + bk);
  v16h a;
#pragma unroll
  for (int e = 0; e < 8; ++e) { a[e] = lo[e]; a[8 + e] = hi[e]; }
  return a;
}

// f16 B fragment from N-major weights (row n holds K contiguous): B[k][n] = W[n*ldk + k]
__device__ __forceinline__ v16h load_b_f16(const _Float16* __restrict__ W, int ldk,
                                           int n0, int k0, int lane) {
  int n  = n0 + (lane & 15);
  int kb = k0 + (lane >> 4) * 8;
  const _Float16* p = W + (size_t)n * ldk + kb;
  v8h lo = *(const v8h*)p;
  v8h hi = *(const v8h*)(p + 16);
  v16h b;
#pragma unroll
  for (int e = 0; e < 8; ++e) { b[e] = lo[e]; b[8 + e] = hi[e]; }
  return b;
}

// B 32x16 from row-major KxN f32 matrix: B[k][n] = W[k*ldn + n]
__device__ __forceinline__ v16h load_b_rm(const float* __restrict__ W, int ldn,
                                          int k0, int n0, int lane) {
  int n  = n0 + (lane & 15);
  int kb = k0 + (lane >> 4) * 8;
  v16h b;
#pragma unroll
  for (int e = 0; e < 16; ++e) b[e] = (_Float16)W[(size_t)(kb + (e & 7) + ((e & 8) ? 16 : 0)) * ldn + n];
  return b;
}

// C/D 16x16 f32: lane col = lane&15; VGPR j holds row j + 8*(lane>>4)
__device__ __forceinline__ void store_tile(float* __restrict__ dst, int ldd,
                                           int m0, int n0, v8f c,
                                           const float* __restrict__ bias,
                                           bool relu, int lane) {
  int col   = n0 + (lane & 15);
  int rbase = m0 + ((lane >> 4) << 3);
  float bv = bias ? bias[col] : 0.0f;
#pragma unroll
  for (int j = 0; j < 8; ++j) {
    float v = c[j] + bv;
    if (relu) v = fmaxf(v, 0.0f);
    dst[(size_t)(rbase + j) * ldd + col] = v;
  }
}

__device__ __forceinline__ float warp_sum(float v) {
#pragma unroll
  for (int o = 16; o > 0; o >>= 1) v += __shfl_xor(v, o, 32);
  return v;
}

// ---------------- generic helpers ----------------
__global__ void zero_kernel(float* p, int n) {
  int i = blockIdx.x * blockDim.x + threadIdx.x;
  if (i < n) p[i] = 0.0f;
}
__global__ void copy_kernel(const float* __restrict__ s, float* __restrict__ d, int n) {
  int i = blockIdx.x * blockDim.x + threadIdx.x;
  if (i < n) d[i] = s[i];
}
__global__ void f32_to_f16_kernel(const float* __restrict__ s, _Float16* __restrict__ d, int n) {
  int i = blockIdx.x * blockDim.x + threadIdx.x;
  if (i < n) d[i] = (_Float16)s[i];
}
// Wo (192x96 row-major) -> N-major f16 (96 rows of 192 contiguous K)
__global__ void wo_transpose_kernel(const float* __restrict__ WoW, _Float16* __restrict__ wo16) {
  int i = blockIdx.x * blockDim.x + threadIdx.x; // 96*192
  if (i >= 96 * 192) return;
  int n = i / 192, k = i % 192;
  wo16[(size_t)n * 192 + k] = (_Float16)WoW[(size_t)k * 96 + n];
}

// ---------------- stage 1: relation attention + sem + h = [feat, sem] ----------------
__global__ void sem_kernel(const float* __restrict__ feat, const int* __restrict__ out_rels,
                           const int* __restrict__ in_rels, const int* __restrict__ rlabn,
                           const float* __restrict__ rel_emb, const float* __restrict__ Wre,
                           const float* __restrict__ bre, float* __restrict__ h) {
  int warp = blockIdx.x * (blockDim.x >> 5) + (threadIdx.x >> 5);
  int lane = threadIdx.x & 31;
  if (warp >= NNODE) return;
  int n = warp;
  float tgt = rel_emb[(size_t)rlabn[n] * INPD + lane];
  float eo[KNEI], ei[KNEI], so[KNEI], si[KNEI];
#pragma unroll
  for (int k = 0; k < KNEI; ++k) {
    eo[k] = rel_emb[(size_t)out_rels[n * KNEI + k] * INPD + lane];
    ei[k] = rel_emb[(size_t)in_rels[n * KNEI + k] * INPD + lane];
    so[k] = warp_sum(eo[k] * tgt);
    si[k] = warp_sum(ei[k] * tgt);
  }
  float mo = so[0], mi = si[0];
#pragma unroll
  for (int k = 1; k < KNEI; ++k) { mo = fmaxf(mo, so[k]); mi = fmaxf(mi, si[k]); }
  float zo = 0.f, zi = 0.f;
#pragma unroll
  for (int k = 0; k < KNEI; ++k) {
    so[k] = __expf(so[k] - mo); zo += so[k];
    si[k] = __expf(si[k] - mi); zi += si[k];
  }
  float osem = 0.f, isem = 0.f;
#pragma unroll
  for (int k = 0; k < KNEI; ++k) { osem += (so[k] / zo) * eo[k]; isem += (si[k] / zi) * ei[k]; }
  float myval = 0.f;
#pragma unroll
  for (int j = 0; j < 16; ++j) {
    float p = osem * Wre[lane * 16 + j] + isem * Wre[(32 + lane) * 16 + j];
    float s = warp_sum(p);
    if (lane == j) myval = 1.0f / (1.0f + __expf(-(s + bre[j])));
  }
  int srcl = (lane >= 16) ? (lane - 16) : 0;
  float sv = __shfl(myval, srcl, 32);
  float hv = (lane < FEATD) ? feat[(size_t)n * FEATD + lane] : sv;
  h[(size_t)n * INPD + lane] = hv;
}

// ---------------- degree ----------------
__global__ void deg_count_kernel(const int* __restrict__ dst, float* __restrict__ deg) {
  int i = blockIdx.x * blockDim.x + threadIdx.x;
  if (i < NEDGE) atomicAdd(&deg[dst[i]], 1.0f);
}
__global__ void rdeg_kernel(float* deg) {
  int i = blockIdx.x * blockDim.x + threadIdx.x;
  if (i < NNODE) deg[i] = 1.0f / fmaxf(deg[i], 1.0f);
}

// ---------------- GNN message + segment-sum, per-graph LDS accumulation ----------------
// Edges are block-diagonal: graph g owns edges [g*2048, (g+1)*2048), nodes [g*128, g*128+128).
// Accumulate the 128x32 aggregate tile in LDS (16KB) with ds-atomics, then stream out.
__global__ void msg_agg_lds_kernel(const float* __restrict__ h, const float* __restrict__ r,
                                   const int* __restrict__ src, const int* __restrict__ dst,
                                   const int* __restrict__ etype, float* __restrict__ agg) {
  __shared__ float acc[NSEQ * INPD]; // 128*32 floats = 16 KB
  int g = blockIdx.x;
  int tid = threadIdx.x;             // 256 threads
  for (int i = tid; i < NSEQ * INPD; i += 256) acc[i] = 0.0f;
  __syncthreads();
  int ebase = g * EPG;
  // work item = (local edge, eighth): 2048*8 items, 4 dims each
  for (int w = tid; w < EPG * 8; w += 256) {
    int el = w >> 3, q = (w & 7) * 4;
    int e = ebase + el;
    int s = src[e];
    int dl = dst[e] & (NSEQ - 1);    // local node id within graph
    int et = etype[e];
    const float* hs = h + (size_t)s * INPD + q;
    const float* re = r + (size_t)et * INPD + q;
#pragma unroll
    for (int i = 0; i < 4; ++i) atomicAdd(&acc[dl * INPD + q + i], hs[i] * re[i]);
  }
  __syncthreads();
  float* out = agg + (size_t)g * NSEQ * INPD;
  for (int i = tid; i < NSEQ * INPD; i += 256) out[i] = acc[i];
}

// ---------------- GNN layer update (WMMA): h' = relu(agg/deg @ Wmsg + h @ Wself + b) ----------------
__global__ void gnn_update_kernel(const float* __restrict__ agg, const float* __restrict__ rdeg,
                                  const float* __restrict__ h, const float* __restrict__ Wmsg,
                                  const float* __restrict__ Wself, const float* __restrict__ bias,
                                  float* __restrict__ hout, float* __restrict__ reprcol) {
  int warp = blockIdx.x * (blockDim.x >> 5) + (threadIdx.x >> 5);
  int lane = threadIdx.x & 31;
  int total = (NNODE / 16) * 2;
  if (warp >= total) return;
  int m0 = (warp >> 1) * 16;
  int n0 = (warp & 1) * 16;
  v8f c = {};
  v16h a0 = load_a_f32(agg, INPD, m0, 0, rdeg, lane);
  v16h b0 = load_b_rm(Wmsg, INPD, 0, n0, lane);
  c = __builtin_amdgcn_wmma_f32_16x16x32_f16(false, a0, false, b0, (short)0, c, false, false);
  v16h a1 = load_a_f32(h, INPD, m0, 0, nullptr, lane);
  v16h b1 = load_b_rm(Wself, INPD, 0, n0, lane);
  c = __builtin_amdgcn_wmma_f32_16x16x32_f16(false, a1, false, b1, (short)0, c, false, false);
  int col = n0 + (lane & 15);
  int rbase = m0 + ((lane >> 4) << 3);
  float bv = bias[col];
#pragma unroll
  for (int j = 0; j < 8; ++j) {
    float v = fmaxf(c[j] + bv, 0.0f);
    hout[(size_t)(rbase + j) * INPD + col] = v;
    reprcol[(size_t)(rbase + j) * OUTD + col] = v;
  }
}

// ---------------- relation update: r' = r @ Wrel ----------------
__global__ void rel_update_kernel(const float* __restrict__ r, const float* __restrict__ W,
                                  float* __restrict__ r2) {
  int idx = blockIdx.x * blockDim.x + threadIdx.x;
  if (idx >= NRREL * INPD) return;
  int i = idx >> 5, j = idx & 31;
  float s = 0.f;
#pragma unroll 8
  for (int k = 0; k < INPD; ++k) s += r[(size_t)i * INPD + k] * W[(size_t)k * INPD + j];
  r2[idx] = s;
}

// ---------------- message (time-major, f16) and h0 ----------------
__global__ void message_kernel(const float* __restrict__ repr, const float* __restrict__ gbias,
                               _Float16* __restrict__ msg16) {
  int idx = blockIdx.x * blockDim.x + threadIdx.x; // NN*96
  if (idx >= NNODE * OUTD) return;
  int j = idx % OUTD;
  int i = idx / OUTD;          // i = b*N + t
  int b = i >> 7, t = i & 127;
  msg16[((size_t)t * BATCH + b) * OUTD + j] = (_Float16)fmaxf(repr[idx] + gbias[j], 0.0f);
}
__global__ void h0_kernel(const float* __restrict__ repr, float* __restrict__ h0) {
  int idx = blockIdx.x * blockDim.x + threadIdx.x; // B*96
  if (idx >= BATCH * OUTD) return;
  int b = idx / OUTD, j = idx % OUTD;
  const float* base = repr + ((size_t)b * NSEQ) * OUTD + j;
  float m = base[0];
  for (int t = 1; t < NSEQ; ++t) m = fmaxf(m, base[(size_t)t * OUTD]);
  h0[idx] = m;
}
__global__ void seth_kernel(const float* __restrict__ h0, float* __restrict__ hst,
                            _Float16* __restrict__ hst16) {
  int i = blockIdx.x * blockDim.x + threadIdx.x;
  if (i >= BATCH * OUTD) return;
  float v = h0[i];
  hst[i] = v;
  hst16[i] = (_Float16)v;
}

// ---------------- gi GEMM with A reuse: gi[NN x 288] = msg16 @ Wih^T + bih ----------------
__global__ void gi_rows_kernel(const _Float16* __restrict__ X, const _Float16* __restrict__ Wh,
                               const float* __restrict__ bias, float* __restrict__ Y) {
  int warp = blockIdx.x * (blockDim.x >> 5) + (threadIdx.x >> 5);
  int lane = threadIdx.x & 31;
  if (warp >= NNODE / 16) return;
  int m0 = warp << 4;
  v16h af[3];
#pragma unroll
  for (int kk = 0; kk < 3; ++kk) af[kk] = load_a_f16(X, OUTD, m0, kk * 32, lane);
  for (int n0 = 0; n0 < 3 * OUTD; n0 += 16) {
    v8f c = {};
#pragma unroll
    for (int kk = 0; kk < 3; ++kk) {
      v16h b = load_b_f16(Wh, OUTD, n0, kk * 32, lane);
      c = __builtin_amdgcn_wmma_f32_16x16x32_f16(false, af[kk], false, b, (short)0, c, false, false);
    }
    store_tile(Y, 3 * OUTD, m0, n0, c, bias, false, lane);
  }
}

// ---------------- recurrent step GEMM: gh[512 x 288] = hst16 @ Whh^T + bhh ----------------
__global__ void gh_step_kernel(const _Float16* __restrict__ hst16, const _Float16* __restrict__ Wh,
                               const float* __restrict__ bhh, float* __restrict__ gh) {
  int warp = blockIdx.x * (blockDim.x >> 5) + (threadIdx.x >> 5);
  int lane = threadIdx.x & 31;
  const int nt = (3 * OUTD) / 16; // 18
  int total = (BATCH / 16) * nt;  // 576
  if (warp >= total) return;
  int m0 = (warp / nt) << 4;
  int n0 = (warp % nt) << 4;
  v8f c = {};
#pragma unroll
  for (int kk = 0; kk < 3; ++kk) {
    v16h a = load_a_f16(hst16, OUTD, m0, kk * 32, lane);
    v16h b = load_b_f16(Wh, OUTD, n0, kk * 32, lane);
    c = __builtin_amdgcn_wmma_f32_16x16x32_f16(false, a, false, b, (short)0, c, false, false);
  }
  store_tile(gh, 3 * OUTD, m0, n0, c, bhh, false, lane);
}

// ---------------- GRU gate elementwise, one time step ----------------
__global__ void gru_gate_kernel(const float* __restrict__ gi, const float* __restrict__ gh,
                                float* __restrict__ hstate, _Float16* __restrict__ hst16,
                                _Float16* __restrict__ ys16_t) {
  int idx = blockIdx.x * blockDim.x + threadIdx.x; // B*96
  if (idx >= BATCH * OUTD) return;
  int b = idx / OUTD, j = idx % OUTD;
  size_t base = (size_t)b * 3 * OUTD;
  float ir = gi[base + j], iz = gi[base + OUTD + j], in_ = gi[base + 2 * OUTD + j];
  float hr = gh[base + j], hz = gh[base + OUTD + j], hn = gh[base + 2 * OUTD + j];
  float r = 1.0f / (1.0f + __expf(-(ir + hr)));
  float z = 1.0f / (1.0f + __expf(-(iz + hz)));
  float n = tanhf(in_ + r * hn);
  float hp = hstate[idx];
  float h2 = (1.0f - z) * n + z * hp;
  hstate[idx] = h2;
  hst16[idx] = (_Float16)h2;
  ys16_t[idx] = (_Float16)fmaxf(h2, 0.0f);
}

// ---------------- Wo GEMM with A reuse + fused concat: nh = relu([ysf,ysb] @ Wo + b) ----------------
__global__ void wo_rows_kernel(const _Float16* __restrict__ ysf16, const _Float16* __restrict__ ysb16,
                               const _Float16* __restrict__ wo16, const float* __restrict__ Wob,
                               float* __restrict__ nh) {
  int warp = blockIdx.x * (blockDim.x >> 5) + (threadIdx.x >> 5);
  int lane = threadIdx.x & 31;
  if (warp >= NNODE / 16) return;
  int m0 = warp << 4;
  int m = m0 + (lane & 15);
  int b_ = m >> 7, t = m & 127;
  const _Float16* rowf = ysf16 + ((size_t)t * BATCH + b_) * OUTD;
  const _Float16* rowb = ysb16 + ((size_t)t * BATCH + b_) * OUTD;
  int bk = (lane >> 4) * 8;
  v16h af[6];
#pragma unroll
  for (int kk = 0; kk < 6; ++kk) {
    const _Float16* row = (kk < 3) ? (rowf + kk * 32) : (rowb + (kk - 3) * 32);
    v8h lo = *(const v8h*)(row + bk);
    v8h hi = *(const v8h*)(row + 16 + bk);
#pragma unroll
    for (int e = 0; e < 8; ++e) { af[kk][e] = lo[e]; af[kk][8 + e] = hi[e]; }
  }
  for (int n0 = 0; n0 < OUTD; n0 += 16) {
    v8f c = {};
#pragma unroll
    for (int kk = 0; kk < 6; ++kk) {
      v16h b = load_b_f16(wo16, 2 * OUTD, n0, kk * 32, lane);
      c = __builtin_amdgcn_wmma_f32_16x16x32_f16(false, af[kk], false, b, (short)0, c, false, false);
    }
    store_tile(nh, OUTD, m0, n0, c, Wob, true, lane);
  }
}

// ---------------- g_out mean + head/tail into g_rep ----------------
__global__ void gout_kernel(const float* __restrict__ nh, float* __restrict__ grep) {
  int idx = blockIdx.x * blockDim.x + threadIdx.x; // B*96
  if (idx >= BATCH * OUTD) return;
  int b = idx / OUTD, j = idx % OUTD;
  const float* base = nh + ((size_t)b * NSEQ) * OUTD + j;
  float s = 0.f;
  for (int t = 0; t < NSEQ; ++t) s += base[(size_t)t * OUTD];
  float* g = grep + (size_t)b * GREPD;
  g[j] = s * (1.0f / (float)NSEQ);
  g[OUTD + j] = base[0];
  g[2 * OUTD + j] = base[OUTD]; // t=1
}
__global__ void rlab_kernel(const int* __restrict__ rlab, const float* __restrict__ r,
                            float* __restrict__ grep) {
  int idx = blockIdx.x * blockDim.x + threadIdx.x; // B*32
  if (idx >= BATCH * EMBD) return;
  int b = idx >> 5, e = idx & 31;
  grep[(size_t)b * GREPD + 3 * OUTD + e] = r[(size_t)rlab[b] * INPD + e];
}

// ---------------- path GRU (3 steps, hidden 32), one thread per (b,p) ----------------
__global__ void path_kernel(const int* __restrict__ in_rels, const int* __restrict__ rlab,
                            const float* __restrict__ r_out,
                            const float* __restrict__ Wih, const float* __restrict__ Whh,
                            const float* __restrict__ bih, const float* __restrict__ bhh,
                            float* __restrict__ last) {
  int idx = blockIdx.x * blockDim.x + threadIdx.x; // B*64
  if (idx >= BATCH * PPATH) return;
  int b = idx >> 6, p = idx & 63;
  int steps[3];
  steps[0] = in_rels[(size_t)b * NSEQ * KNEI + (p >> 3)];        // heads_rels[b][p/K]
  steps[1] = rlab[b];
  steps[2] = in_rels[(size_t)b * NSEQ * KNEI + KNEI + (p & 7)];  // tails_rels[b][p%K]
  float hcur[EMBD];
#pragma unroll
  for (int e = 0; e < EMBD; ++e) hcur[e] = 0.0f;
  for (int s = 0; s < 3; ++s) {
    const float* x = r_out + (size_t)steps[s] * INPD;
    float hnew[EMBD];
#pragma unroll 1
    for (int j = 0; j < EMBD; ++j) {
      float gir = bih[j], giz = bih[EMBD + j], gin = bih[2 * EMBD + j];
      float ghr = bhh[j], ghz = bhh[EMBD + j], ghn = bhh[2 * EMBD + j];
#pragma unroll 4
      for (int k = 0; k < EMBD; ++k) {
        float xk = x[k], hk = hcur[k];
        gir += xk * Wih[j * EMBD + k];
        giz += xk * Wih[(EMBD + j) * EMBD + k];
        gin += xk * Wih[(2 * EMBD + j) * EMBD + k];
        ghr += hk * Whh[j * EMBD + k];
        ghz += hk * Whh[(EMBD + j) * EMBD + k];
        ghn += hk * Whh[(2 * EMBD + j) * EMBD + k];
      }
      float rr = 1.0f / (1.0f + __expf(-(gir + ghr)));
      float zz = 1.0f / (1.0f + __expf(-(giz + ghz)));
      float nn = tanhf(gin + rr * ghn);
      hnew[j] = (1.0f - zz) * nn + zz * hcur[j];
    }
#pragma unroll
    for (int j = 0; j < EMBD; ++j) hcur[j] = hnew[j];
  }
#pragma unroll
  for (int j = 0; j < EMBD; ++j) last[(size_t)idx * EMBD + j] = hcur[j];
}

// ---------------- path attention + g_p ----------------
__global__ void path_att_kernel(const float* __restrict__ last, float* __restrict__ grep) {
  __shared__ float sc[PPATH];
  __shared__ float att[PPATH];
  int b = blockIdx.x, p = threadIdx.x;
  const float* rl = grep + (size_t)b * GREPD + 3 * OUTD; // r_lab_e
  const float* lp = last + ((size_t)b * PPATH + p) * EMBD;
  float d = 0.f;
#pragma unroll 8
  for (int e = 0; e < EMBD; ++e) d += lp[e] * rl[e];
  sc[p] = d;
  __syncthreads();
  float m = -1e30f;
  for (int q = 0; q < PPATH; ++q) m = fmaxf(m, sc[q]);
  float ex = __expf(d - m);
  att[p] = ex;
  __syncthreads();
  float sum = 0.f;
  for (int q = 0; q < PPATH; ++q) sum += att[q];
  float a = ex / sum;
  att[p] = a;
  __syncthreads();
  if (p < EMBD) {
    float g = 0.f;
    for (int q = 0; q < PPATH; ++q) g += att[q] * last[((size_t)b * PPATH + q) * EMBD + p];
    grep[(size_t)b * GREPD + 3 * OUTD + EMBD + p] = g;
  }
}

// ---------------- final FC: out[b] = dot(g_rep[b], fc_W) + fc_b ----------------
__global__ void fc_kernel(const float* __restrict__ grep, const float* __restrict__ fcW,
                          const float* __restrict__ fcb, float* __restrict__ out) {
  int warp = blockIdx.x * (blockDim.x >> 5) + (threadIdx.x >> 5);
  int lane = threadIdx.x & 31;
  if (warp >= BATCH) return;
  float s = 0.f;
  for (int c = lane; c < GREPD; c += 32) s += grep[(size_t)warp * GREPD + c] * fcW[c];
  s = warp_sum(s);
  if (lane == 0) out[warp] = s + fcb[0];
}

// ================= host orchestration =================
static inline int cdiv(int a, int b) { return (a + b - 1) / b; }

extern "C" void kernel_launch(void* const* d_in, const int* in_sizes, int n_in,
                              void* d_out, int out_size, void* d_ws, size_t ws_size,
                              hipStream_t stream) {
  (void)in_sizes; (void)n_in; (void)out_size; (void)ws_size;
  const float* feat     = (const float*)d_in[0];
  const int*   out_rels = (const int*)d_in[1];
  const int*   in_rels  = (const int*)d_in[2];
  const int*   rlabn    = (const int*)d_in[3];
  const int*   srcI     = (const int*)d_in[4];
  const int*   dstI     = (const int*)d_in[5];
  const int*   etype    = (const int*)d_in[6];
  const int*   rlab     = (const int*)d_in[7];
  const float* rel_emb  = (const float*)d_in[8];
  const float* Wre      = (const float*)d_in[9];
  const float* bre      = (const float*)d_in[10];
  const float* Wmsg     = (const float*)d_in[11];
  const float* Wself    = (const float*)d_in[12];
  const float* bgnn     = (const float*)d_in[13];
  const float* Wrel     = (const float*)d_in[14];
  const float* gbias    = (const float*)d_in[15];
  const float* Wih_f    = (const float*)d_in[16];
  const float* Whh_f    = (const float*)d_in[17];
  const float* bih_f    = (const float*)d_in[18];
  const float* bhh_f    = (const float*)d_in[19];
  const float* Wih_b    = (const float*)d_in[20];
  const float* Whh_b    = (const float*)d_in[21];
  const float* bih_b    = (const float*)d_in[22];
  const float* bhh_b    = (const float*)d_in[23];
  const float* WoW      = (const float*)d_in[24];
  const float* Wob      = (const float*)d_in[25];
  const float* pWih     = (const float*)d_in[26];
  const float* pWhh     = (const float*)d_in[27];
  const float* pbih     = (const float*)d_in[28];
  const float* pbhh     = (const float*)d_in[29];
  const float* fcW      = (const float*)d_in[30];
  const float* fcb      = (const float*)d_in[31];
  float* out = (float*)d_out;
  float* ws  = (float*)d_ws;

  float* h_a  = ws + OFF_HA;
  float* h_b  = ws + OFF_HB;
  float* agg  = ws + OFF_AGG;
  float* deg  = ws + OFF_DEG;
  float* r_a  = ws + OFF_RA;
  float* r_b  = ws + OFF_RB;
  float* repr = ws + OFF_REPR;
  float* gi   = ws + OFF_GI;
  float* h0b  = ws + OFF_H0;
  float* hst  = ws + OFF_HST;
  float* gh   = ws + OFF_GH;
  float* grep = ws + OFF_GREP;
  _Float16* msg16 = (_Float16*)(ws + OFF_HA);   // aliases h_a..agg after GNN loop
  _Float16* ysf16 = (_Float16*)(ws + OFF_YSF16);
  _Float16* ysb16 = (_Float16*)(ws + OFF_YSB16);
  _Float16* hst16 = (_Float16*)(ws + OFF_HST16);
  _Float16* wih16 = (_Float16*)(ws + OFF_WIH16);
  _Float16* whh16 = (_Float16*)(ws + OFF_WHH16);
  _Float16* wo16  = (_Float16*)(ws + OFF_WO16);
  float* nh    = gi;                            // aliases gi after GRU done
  float* lastb = gi + (size_t)NNODE * OUTD;

  const int TB = 256;

  // degree
  zero_kernel<<<cdiv(NNODE, TB), TB, 0, stream>>>(deg, NNODE);
  deg_count_kernel<<<cdiv(NEDGE, TB), TB, 0, stream>>>(dstI, deg);
  rdeg_kernel<<<cdiv(NNODE, TB), TB, 0, stream>>>(deg);

  // attention + sem + h
  sem_kernel<<<NNODE / 8, TB, 0, stream>>>(feat, out_rels, in_rels, rlabn, rel_emb, Wre, bre, h_a);

  // r init
  copy_kernel<<<cdiv(NRREL * INPD, TB), TB, 0, stream>>>(rel_emb, r_a, NRREL * INPD);

  // GNN layers (LDS-accumulated segment sum: one workgroup per graph)
  float* hc = h_a; float* hn2 = h_b; float* rc = r_a; float* rn = r_b;
  for (int l = 0; l < 3; ++l) {
    msg_agg_lds_kernel<<<BATCH, TB, 0, stream>>>(hc, rc, srcI, dstI, etype, agg);
    gnn_update_kernel<<<(NNODE / 16) * 2 / 8, TB, 0, stream>>>(
        agg, deg, hc, Wmsg + l * INPD * INPD, Wself + l * INPD * INPD, bgnn + l * INPD,
        hn2, repr + l * INPD);
    rel_update_kernel<<<cdiv(NRREL * INPD, TB), TB, 0, stream>>>(rc, Wrel + l * INPD * INPD, rn);
    float* t1 = hc; hc = hn2; hn2 = t1;
    float* t2 = rc; rc = rn; rn = t2;
  }
  // rc = r_emb_out

  // GRU input staging (f16) and h0 (note: message_kernel overwrites the h_a..agg region)
  h0_kernel<<<cdiv(BATCH * OUTD, TB), TB, 0, stream>>>(repr, h0b);
  message_kernel<<<cdiv(NNODE * OUTD, TB), TB, 0, stream>>>(repr, gbias, msg16);

  const int GI_BLOCKS = (NNODE / 16) / 8;            // 512
  const int GH_BLOCKS = ((BATCH / 16) * 18) / 8;     // 72

  // ---- forward GRU ----
  f32_to_f16_kernel<<<cdiv(288 * 96, TB), TB, 0, stream>>>(Wih_f, wih16, 288 * 96);
  f32_to_f16_kernel<<<cdiv(288 * 96, TB), TB, 0, stream>>>(Whh_f, whh16, 288 * 96);
  gi_rows_kernel<<<GI_BLOCKS, TB, 0, stream>>>(msg16, wih16, bih_f, gi);
  seth_kernel<<<cdiv(BATCH * OUTD, TB), TB, 0, stream>>>(h0b, hst, hst16);
  for (int t = 0; t < NSEQ; ++t) {
    gh_step_kernel<<<GH_BLOCKS, TB, 0, stream>>>(hst16, whh16, bhh_f, gh);
    gru_gate_kernel<<<cdiv(BATCH * OUTD, TB), TB, 0, stream>>>(
        gi + (size_t)t * BATCH * 3 * OUTD, gh, hst, hst16, ysf16 + (size_t)t * BATCH * OUTD);
  }
  // ---- backward GRU (store un-reversed so concat aligns by t) ----
  f32_to_f16_kernel<<<cdiv(288 * 96, TB), TB, 0, stream>>>(Wih_b, wih16, 288 * 96);
  f32_to_f16_kernel<<<cdiv(288 * 96, TB), TB, 0, stream>>>(Whh_b, whh16, 288 * 96);
  gi_rows_kernel<<<GI_BLOCKS, TB, 0, stream>>>(msg16, wih16, bih_b, gi);
  seth_kernel<<<cdiv(BATCH * OUTD, TB), TB, 0, stream>>>(h0b, hst, hst16);
  for (int s = 0; s < NSEQ; ++s) {
    int t = NSEQ - 1 - s;
    gh_step_kernel<<<GH_BLOCKS, TB, 0, stream>>>(hst16, whh16, bhh_b, gh);
    gru_gate_kernel<<<cdiv(BATCH * OUTD, TB), TB, 0, stream>>>(
        gi + (size_t)t * BATCH * 3 * OUTD, gh, hst, hst16, ysb16 + (size_t)t * BATCH * OUTD);
  }

  // node_hidden = relu(relu(concat(ysf,ysb)) @ Wo + b)   (nh aliases gi)
  wo_transpose_kernel<<<cdiv(96 * 192, TB), TB, 0, stream>>>(WoW, wo16);
  wo_rows_kernel<<<GI_BLOCKS, TB, 0, stream>>>(ysf16, ysb16, wo16, Wob, nh);

  // g_rep assembly
  gout_kernel<<<cdiv(BATCH * OUTD, TB), TB, 0, stream>>>(nh, grep);
  rlab_kernel<<<cdiv(BATCH * EMBD, TB), TB, 0, stream>>>(rlab, rc, grep);
  path_kernel<<<cdiv(BATCH * PPATH, TB), TB, 0, stream>>>(in_rels, rlab, rc, pWih, pWhh, pbih, pbhh, lastb);
  path_att_kernel<<<BATCH, PPATH, 0, stream>>>(lastb, grep);

  // final FC
  fc_kernel<<<BATCH / 8, TB, 0, stream>>>(grep, fcW, fcb, out);
}
